// rbfNet_72138270704369
// MI455X (gfx1250) — matrix operand order repaired and checked by reading, not compiled, not run on hardware
//
#include <hip/hip_runtime.h>
#include <hip/hip_bf16.h>

// ---------------------------------------------------------------------------
// RBF network forward for MI455X (gfx1250, wave32, WMMA, async-LDS DMA).
//   sq[b,o]  = ||x_b||^2 + ||c_o||^2 - 2 * <x_b, c_o>        (bf16 WMMA GEMM)
//   basis    = exp(-max(sq,0) * exp(-2*log_sigma_o))
//   out[b,n] = sum_o basis[b,o] * fc_w[n,o] + fc_b[n]
// ---------------------------------------------------------------------------

typedef __bf16 bf16_t;
typedef __attribute__((ext_vector_type(16))) __bf16 v16bf;
typedef __attribute__((ext_vector_type(8)))  __bf16 v8bf;
typedef __attribute__((ext_vector_type(4)))  __bf16 v4bf;
typedef __attribute__((ext_vector_type(8)))  float  v8f;
typedef __attribute__((ext_vector_type(4)))  float  v4f;
typedef __attribute__((ext_vector_type(4)))  int    v4i;

#define RBF_B   16384
#define RBF_IN  512
#define RBF_O   1024
#define RBF_C   5
#define BM      64
#define BN      32
#define LDA     (RBF_IN + 8)   // padded LDS row stride (bf16): conflict-free b128
#define LDB     (RBF_IN + 8)
#define KFRAG   (RBF_IN / 32)  // 16 WMMA steps per tile

#if defined(__AMDGCN__) && __has_builtin(__builtin_amdgcn_global_load_async_to_lds_b128)
#define RBF_ASYNC 1
#else
#define RBF_ASYNC 0
#endif

#if RBF_ASYNC
typedef __attribute__((address_space(1))) v4i* gv4i_p;   // global b128 payload
typedef __attribute__((address_space(3))) v4i* lv4i_p;   // LDS b128 payload

__device__ __forceinline__ void async_b128(const bf16_t* g, bf16_t* l) {
    __builtin_amdgcn_global_load_async_to_lds_b128((gv4i_p)g, (lv4i_p)l, 0, 0);
}
__device__ __forceinline__ void wait_async0() {
#if __has_builtin(__builtin_amdgcn_s_wait_asynccnt)
    __builtin_amdgcn_s_wait_asynccnt(0);
#else
    asm volatile("s_wait_asynccnt 0x0" ::: "memory");
#endif
}
#endif

// ---------------------------------------------------------------------------
// Kernel 1: f32 -> bf16 conversion + row sum-of-squares.
// One block (128 threads) per row; rows [0, B) are inputs, [B, B+O) centres.
// ---------------------------------------------------------------------------
__global__ __launch_bounds__(128) void rbf_convert(
    const float* __restrict__ x, const float* __restrict__ c,
    bf16_t* __restrict__ xb, bf16_t* __restrict__ cb,
    float* __restrict__ x2, float* __restrict__ c2)
{
    const int row = blockIdx.x;
    const float* src;
    bf16_t* dst;
    float* sdst;
    if (row < RBF_B) {
        src  = x  + (size_t)row * RBF_IN;
        dst  = xb + (size_t)row * RBF_IN;
        sdst = x2 + row;
    } else {
        const int r = row - RBF_B;
        src  = c  + (size_t)r * RBF_IN;
        dst  = cb + (size_t)r * RBF_IN;
        sdst = c2 + r;
    }
    const int t = threadIdx.x;              // 128 threads * 4 floats = 512
    v4f v = ((const v4f*)src)[t];
    ((v4bf*)dst)[t] = __builtin_convertvector(v, v4bf);
    float s = v.x * v.x + v.y * v.y + v.z * v.z + v.w * v.w;
    #pragma unroll
    for (int m = 16; m >= 1; m >>= 1) s += __shfl_xor(s, m, 32);
    __shared__ float red[4];
    if ((t & 31) == 0) red[t >> 5] = s;
    __syncthreads();
    if (t == 0) *sdst = red[0] + red[1] + red[2] + red[3];
}

// ---------------------------------------------------------------------------
// Kernel 2: fused distance-GEMM + RBF + 5-class projection.
// 256 blocks x 256 threads (8 wave32s). Each block owns 64 rows of B,
// sweeps all 1024 centres in 32-wide N tiles. A tile + double-buffered B
// tiles live in LDS; B tiles are DMA'd with global_load_async_to_lds_b128
// one tile ahead of the WMMA burst (ASYNCcnt pipeline).
// ---------------------------------------------------------------------------
__global__ __launch_bounds__(256) void rbf_main(
    const bf16_t* __restrict__ xb, const bf16_t* __restrict__ cb,
    const float* __restrict__ x2, const float* __restrict__ c2,
    const float* __restrict__ lsig, const float* __restrict__ fcw,
    const float* __restrict__ fcb, float* __restrict__ out)
{
    __shared__ bf16_t As[BM * LDA];            // 64 x 520 bf16 = 66,560 B
#if RBF_ASYNC
    __shared__ bf16_t Bs[2][BN * LDB];         // 2 x 33,280 B (double buffer)
#else
    __shared__ bf16_t Bs[1][BN * LDB];
#endif
    __shared__ float outAcc[2][BM][RBF_C];     // 2,560 B

    const int tid = threadIdx.x;
    const int m0  = blockIdx.x * BM;

    for (int idx = tid; idx < 2 * BM * RBF_C; idx += 256)
        ((float*)outAcc)[idx] = 0.0f;

#if RBF_ASYNC
    // Async-DMA the A tile (64 rows x 512 bf16) and the first B tile.
    for (int idx = tid; idx < BM * (RBF_IN / 8); idx += 256) {
        const int row = idx >> 6, seg = idx & 63;
        async_b128(xb + (size_t)(m0 + row) * RBF_IN + seg * 8,
                   &As[row * LDA + seg * 8]);
    }
    for (int idx = tid; idx < BN * (RBF_IN / 8); idx += 256) {
        const int col = idx >> 6, seg = idx & 63;
        async_b128(cb + (size_t)col * RBF_IN + seg * 8,
                   &Bs[0][col * LDB + seg * 8]);
    }
    wait_async0();
    __syncthreads();
#else
    for (int idx = tid; idx < BM * (RBF_IN / 8); idx += 256) {
        const int row = idx >> 6, seg = idx & 63;
        *(uint4*)&As[row * LDA + seg * 8] =
            *(const uint4*)(xb + (size_t)(m0 + row) * RBF_IN + seg * 8);
    }
    __syncthreads();
#endif

    const int lane = tid & 31;
    const int wr   = (tid >> 5) & 3;   // 0..3  -> M sub-tile
    const int wc   = tid >> 7;         // 0..1  -> N half
    const int ln   = lane & 15;
    const int hi   = lane >> 4;        // lane half (K-swizzle select)

    // ||x||^2 for the 8 rows this lane's C fragment covers (M = i + 8*hi).
    float x2r[8];
    #pragma unroll
    for (int i = 0; i < 8; ++i) x2r[i] = x2[m0 + 16 * wr + i + 8 * hi];

    float acc[8][RBF_C];
    #pragma unroll
    for (int i = 0; i < 8; ++i)
        #pragma unroll
        for (int cc = 0; cc < RBF_C; ++cc) acc[i][cc] = 0.0f;

    // Hoist all 16 A fragments into VGPRs once (ISA 16-bit A layout: lane
    // half holds K = {k0+8h..+7, k0+16+8h..+7}).
    const bf16_t* arow = &As[(16 * wr + ln) * LDA + 8 * hi];
    v16bf afrag[KFRAG];
    #pragma unroll
    for (int kk = 0; kk < KFRAG; ++kk) {
        v8bf alo = *(const v8bf*)(arow + kk * 32);
        v8bf ahi = *(const v8bf*)(arow + kk * 32 + 16);
        afrag[kk] = __builtin_shufflevector(alo, ahi,
            0, 1, 2, 3, 4, 5, 6, 7, 8, 9, 10, 11, 12, 13, 14, 15);
    }

    for (int n0 = 0; n0 < RBF_O; n0 += BN) {
#if RBF_ASYNC
        const int buf = (n0 >> 5) & 1;
        if (n0) {               // current buffer's DMA (issued last iter) done
            wait_async0();
            __syncthreads();
        }
        if (n0 + BN < RBF_O) {  // prefetch next tile into the other buffer
            const bf16_t* csrc = cb + (size_t)(n0 + BN) * RBF_IN;
            for (int idx = tid; idx < BN * (RBF_IN / 8); idx += 256) {
                const int col = idx >> 6, seg = idx & 63;
                async_b128(csrc + (size_t)col * RBF_IN + seg * 8,
                           &Bs[buf ^ 1][col * LDB + seg * 8]);
            }
        }
#else
        const int buf = 0;
        __syncthreads();
        {
            const bf16_t* csrc = cb + (size_t)n0 * RBF_IN;
            for (int idx = tid; idx < BN * (RBF_IN / 8); idx += 256) {
                const int col = idx >> 6, seg = idx & 63;
                *(uint4*)&Bs[0][col * LDB + seg * 8] =
                    *(const uint4*)(csrc + (size_t)col * RBF_IN + seg * 8);
            }
        }
        __syncthreads();
#endif

        const int n = n0 + 16 * wc + ln;                 // this lane's column
        // B layout: lane half holds K = {k0+8h..+7, k0+16+8h..+7} of column n.
        const bf16_t* brow = &Bs[buf][(16 * wc + ln) * LDB + 8 * hi];

        v8f cfrag = {};
        #pragma unroll
        for (int kk = 0; kk < KFRAG; ++kk) {
            v8bf blo = *(const v8bf*)(brow + kk * 32);
            v8bf bhi = *(const v8bf*)(brow + kk * 32 + 16);
            v16bf b = __builtin_shufflevector(blo, bhi,
                0, 1, 2, 3, 4, 5, 6, 7, 8, 9, 10, 11, 12, 13, 14, 15);
            cfrag = __builtin_amdgcn_wmma_f32_16x16x32_bf16(
                false, afrag[kk], false, b, (short)0, cfrag, false, false);
        }

        // Fused epilogue: sq -> RBF basis -> 5-class partial sums.
        const float c2v = c2[n];
        const float is2 = __expf(-2.0f * lsig[n]);       // 1/sigma^2
        float w[RBF_C];
        #pragma unroll
        for (int cc = 0; cc < RBF_C; ++cc) w[cc] = fcw[cc * RBF_O + n];

        #pragma unroll
        for (int i = 0; i < 8; ++i) {
            float sq  = x2r[i] + c2v - 2.0f * cfrag[i];
            sq        = fmaxf(sq, 0.0f);
            float bas = __expf(-sq * is2);
            #pragma unroll
            for (int cc = 0; cc < RBF_C; ++cc)
                acc[i][cc] = fmaf(bas, w[cc], acc[i][cc]);
        }
    }

    // Reduce over the 16 lanes sharing a row (same half), park in LDS.
    #pragma unroll
    for (int i = 0; i < 8; ++i) {
        #pragma unroll
        for (int cc = 0; cc < RBF_C; ++cc) {
            float v = acc[i][cc];
            v += __shfl_xor(v, 1, 16);
            v += __shfl_xor(v, 2, 16);
            v += __shfl_xor(v, 4, 16);
            v += __shfl_xor(v, 8, 16);
            if (ln == 0) outAcc[wc][16 * wr + i + 8 * hi][cc] = v;
        }
    }
    __syncthreads();

    // Combine the two N-halves, add bias, write final rows (block-exclusive).
    for (int idx = tid; idx < BM * RBF_C; idx += 256) {
        const int r = idx / RBF_C, cc = idx % RBF_C;
        out[(size_t)(m0 + r) * RBF_C + cc] =
            outAcc[0][r][cc] + outAcc[1][r][cc] + fcb[cc];
    }
}

// ---------------------------------------------------------------------------
extern "C" void kernel_launch(void* const* d_in, const int* in_sizes, int n_in,
                              void* d_out, int out_size, void* d_ws, size_t ws_size,
                              hipStream_t stream) {
    const float* inputs  = (const float*)d_in[0];   // [16384, 512]
    const float* centres = (const float*)d_in[1];   // [1024, 512]
    const float* lsig    = (const float*)d_in[2];   // [1024]
    const float* fcw     = (const float*)d_in[3];   // [5, 1024]
    const float* fcb     = (const float*)d_in[4];   // [5]
    float* out           = (float*)d_out;           // [16384, 5]

    // Workspace layout (~17.1 MiB): bf16 inputs, bf16 centres, x2, c2.
    char* ws = (char*)d_ws;
    const size_t xb_bytes = (size_t)RBF_B * RBF_IN * sizeof(bf16_t);  // 16 MiB
    const size_t cb_bytes = (size_t)RBF_O * RBF_IN * sizeof(bf16_t);  //  1 MiB
    bf16_t* xb = (bf16_t*)ws;
    bf16_t* cb = (bf16_t*)(ws + xb_bytes);
    float*  x2 = (float*)(ws + xb_bytes + cb_bytes);
    float*  c2 = x2 + RBF_B;

    rbf_convert<<<RBF_B + RBF_O, 128, 0, stream>>>(inputs, centres, xb, cb, x2, c2);
    rbf_main<<<RBF_B / BM, 256, 0, stream>>>(xb, cb, x2, c2, lsig, fcw, fcb, out);
}
